// MLPF_71880572666452
// MI455X (gfx1250) — matrix operand order repaired and checked by reading, not compiled, not run on hardware
//
#include <hip/hip_runtime.h>
#include <hip/hip_bf16.h>

typedef __attribute__((ext_vector_type(16))) _Float16 v16h;
typedef __attribute__((ext_vector_type(8)))  float    v8f;

#define BPTS 4096
#define NBATCH 8
#define MROWS (NBATCH * BPTS)

// ---------------------------------------------------------------------------
// Fused 4-layer MLP (Linear->ELU x3 -> Linear), fp16 WMMA, fp32 accumulate.
// Block = 128 threads = 4 waves; each wave owns a 16-row slab of a 64-row tile.
// Activations live in LDS (fp16, padded stride), weights staged per-32-k slice.
// ---------------------------------------------------------------------------

template <int NT, int H, bool LAST>
__device__ __forceinline__ void mlp_layer(
    _Float16* act, _Float16* wsl, float* bls,
    const float* __restrict__ Wg, const float* __restrict__ Bg,
    int kact, int nact, int ktn, int tid,
    float* __restrict__ gout, int ostride, int ocol, long row0, int nout)
{
    constexpr int AS   = H + 8;    // padded activation row stride (halves)
    constexpr int NPAD = NT * 16;
    const int lane = tid & 31;
    const int wv   = tid >> 5;
    const int m16  = lane & 15;
    const int hi   = lane >> 4;

    __syncthreads();   // previous layer's act writes / bls reads complete
    for (int n = tid; n < NPAD; n += 128) bls[n] = (n < nact) ? Bg[n] : 0.f;

    v8f acc[NT];
#pragma unroll
    for (int t = 0; t < NT; ++t) {
        v8f z = {0.f, 0.f, 0.f, 0.f, 0.f, 0.f, 0.f, 0.f};
        acc[t] = z;
    }

    for (int kt = 0; kt < ktn; ++kt) {
        __syncthreads();
        // Stage W[kt*32 : kt*32+32, 0:NPAD] transposed into LDS as fp16,
        // wsl[n*36 + k2], zero-padded beyond (kact, nact).
        for (int i = tid; i < NPAD * 32; i += 128) {
            int n  = i >> 5;
            int k2 = i & 31;
            int kg = (kt << 5) + k2;
            float v = (kg < kact && n < nact) ? Wg[(long)kg * nact + n] : 0.f;
            wsl[n * 36 + k2] = (_Float16)v;
        }
        __syncthreads();

        // A fragment: 16x32 fp16, row m = lane&15, K striped per ISA layout.
        union { v16h v; unsigned u[8]; } au;
        const unsigned* arow = (const unsigned*)(act + (wv * 16 + m16) * AS);
#pragma unroll
        for (int v = 0; v < 8; ++v) {
            int k = (kt << 5) + ((v >> 2) << 4) + (hi << 3) + ((v & 3) << 1);
            au.u[v] = arow[k >> 1];
        }

#pragma unroll
        for (int t = 0; t < NT; ++t) {
            // B fragment: 32x16 fp16, col n = lane&15, K striped per ISA layout (same as A).
            union { v16h v; unsigned u[8]; } bu;
            const unsigned* brow =
                (const unsigned*)(wsl + (t * 16 + m16) * 36);
#pragma unroll
            for (int v = 0; v < 8; ++v) bu.u[v] = brow[((v >> 2) << 3) + (hi << 2) + (v & 3)];
            acc[t] = __builtin_amdgcn_wmma_f32_16x16x32_f16(
                false, au.v, false, bu.v, (short)0, acc[t], false, false);
        }
    }
    __syncthreads();   // all act reads done before overwrite

#pragma unroll
    for (int t = 0; t < NT; ++t) {
        int col  = t * 16 + m16;
        float bb = bls[col];
#pragma unroll
        for (int r = 0; r < 8; ++r) {
            int mm    = wv * 16 + r + (hi << 3);
            float val = acc[t][r] + bb;
            if constexpr (!LAST) {
                val = (val > 0.f) ? val : (__expf(val) - 1.f);   // ELU
                act[mm * AS + col] = (_Float16)val;
            } else {
                if (col < nout)
                    gout[(row0 + mm) * (long)ostride + ocol + col] = val;
            }
        }
    }
}

template <int KPAD0, int H, int NTH, int NTOUT>
__global__ __launch_bounds__(128) void fused_mlp_kernel(
    const float* __restrict__ in0, int w0, int s0,
    const float* __restrict__ in1, int w1, int s1,
    const float* __restrict__ W0, const float* __restrict__ B0,
    const float* __restrict__ W1, const float* __restrict__ B1,
    const float* __restrict__ W2, const float* __restrict__ B2,
    const float* __restrict__ W3, const float* __restrict__ B3,
    int hact, int nout,
    float* __restrict__ gout, int ostride, int ocol)
{
    constexpr int AS = H + 8;
    __shared__ _Float16 act[64 * AS];
    __shared__ _Float16 wsl[H * 36];
    __shared__ float    bls[H];

    const int  tid  = threadIdx.x;
    const long row0 = (long)blockIdx.x * 64;

    // Stage (optionally concatenated) input, fp32 -> fp16, zero-padded to KPAD0.
    for (int i = tid; i < 64 * KPAD0; i += 128) {
        int  r  = i / KPAD0;
        int  c  = i % KPAD0;
        long gr = row0 + r;
        float v = 0.f;
        if (c < w0)            v = in0[gr * (long)s0 + c];
        else if (c < w0 + w1)  v = in1[gr * (long)s1 + (c - w0)];
        act[r * AS + c] = (_Float16)v;
    }

    mlp_layer<NTH, H, false>(act, wsl, bls, W0, B0, w0 + w1, hact, KPAD0 / 32,
                             tid, nullptr, 0, 0, 0, 0);
    mlp_layer<NTH, H, false>(act, wsl, bls, W1, B1, hact, hact, H / 32,
                             tid, nullptr, 0, 0, 0, 0);
    mlp_layer<NTH, H, false>(act, wsl, bls, W2, B2, hact, hact, H / 32,
                             tid, nullptr, 0, 0, 0, 0);
    mlp_layer<NTOUT, H, true>(act, wsl, bls, W3, B3, hact, nout, H / 32,
                              tid, gout, ostride, ocol, row0, nout);
}

// ---------------------------------------------------------------------------
// GravNet: s/h/|s|^2 projections (tiny GEMVs).
// ---------------------------------------------------------------------------
__global__ __launch_bounds__(256) void gravnet_prep_kernel(
    const float* __restrict__ emb,
    const float* __restrict__ Ws, const float* __restrict__ bs,
    const float* __restrict__ Wh, const float* __restrict__ bh,
    float* __restrict__ s, float* __restrict__ h, float* __restrict__ sq)
{
    __shared__ float lws[32 * 4], lbs[4], lwh[32 * 8], lbh[8];
    const int tid = threadIdx.x;
    if (tid < 128) lws[tid] = Ws[tid];
    if (tid < 4)   lbs[tid] = bs[tid];
    if (tid < 8)   lbh[tid] = bh[tid];
    if (tid < 256) lwh[tid] = Wh[tid];
    __syncthreads();

    const long p = (long)blockIdx.x * 256 + tid;
    float e[32];
#pragma unroll
    for (int k = 0; k < 32; ++k) e[k] = emb[p * 32 + k];

    float acc2 = 0.f;
#pragma unroll
    for (int j = 0; j < 4; ++j) {
        float a = lbs[j];
#pragma unroll
        for (int k = 0; k < 32; ++k) a += e[k] * lws[k * 4 + j];
        s[p * 4 + j] = a;
        acc2 += a * a;
    }
    sq[p] = acc2;
#pragma unroll
    for (int j = 0; j < 8; ++j) {
        float a = lbh[j];
#pragma unroll
        for (int k = 0; k < 32; ++k) a += e[k] * lwh[k * 8 + j];
        h[p * 8 + j] = a;
    }
}

// ---------------------------------------------------------------------------
// GravNet: per-thread query kNN (K=4) scan over 4096 candidates of its batch,
// edge weights exp(-10 d^2), mean+max aggregation, fused output projection.
// ---------------------------------------------------------------------------
__global__ __launch_bounds__(128) void gravnet_knn_kernel(
    const float* __restrict__ emb,
    const float* __restrict__ s, const float* __restrict__ h,
    const float* __restrict__ sq,
    const float* __restrict__ Wo1, const float* __restrict__ Wo2,
    const float* __restrict__ bo2,
    float* __restrict__ embo)
{
    __shared__ float sch[128 * 4];
    __shared__ float sqch[128];
    __shared__ float lw1[32 * 32];
    __shared__ float lw2[16 * 32];
    __shared__ float lb2[32];

    const int  tid  = threadIdx.x;
    const long base = (long)blockIdx.y * BPTS;
    const long q    = base + blockIdx.x * 128 + tid;

    for (int i = tid; i < 1024; i += 128) lw1[i] = Wo1[i];
    for (int i = tid; i < 512;  i += 128) lw2[i] = Wo2[i];
    if (tid < 32) lb2[tid] = bo2[tid];

    float sv[4];
#pragma unroll
    for (int j = 0; j < 4; ++j) sv[j] = s[q * 4 + j];
    const float sqv = sq[q];

    float bd[4] = {3.0e38f, 3.0e38f, 3.0e38f, 3.0e38f};
    int   bi[4] = {0, 0, 0, 0};

    for (int c0 = 0; c0 < BPTS; c0 += 128) {
        __syncthreads();
        const long ci = base + c0 + tid;
#pragma unroll
        for (int j = 0; j < 4; ++j) sch[tid * 4 + j] = s[ci * 4 + j];
        sqch[tid] = sq[ci];
        __syncthreads();

        for (int j = 0; j < 128; ++j) {
            float dot = sv[0] * sch[j * 4 + 0] + sv[1] * sch[j * 4 + 1] +
                        sv[2] * sch[j * 4 + 2] + sv[3] * sch[j * 4 + 3];
            float d = sqv + sqch[j] - 2.f * dot;
            if (d < bd[3]) {
                float dd = d; int ii = c0 + j;
#pragma unroll
                for (int k = 0; k < 4; ++k) {
                    if (dd < bd[k]) {
                        float td = bd[k]; int ti = bi[k];
                        bd[k] = dd; bi[k] = ii;
                        dd = td; ii = ti;
                    }
                }
            }
        }
    }

    float mean[8], mx[8];
#pragma unroll
    for (int j = 0; j < 8; ++j) { mean[j] = 0.f; mx[j] = -3.0e38f; }
#pragma unroll
    for (int k = 0; k < 4; ++k) {
        float d2 = bd[k] > 0.f ? bd[k] : 0.f;   // clamp numeric negatives
        float w  = __expf(-10.f * d2);
        const float* hp = &h[(base + bi[k]) * 8];
#pragma unroll
        for (int j = 0; j < 8; ++j) {
            float m = hp[j] * w;
            mean[j] += m;
            mx[j] = fmaxf(mx[j], m);
        }
    }
#pragma unroll
    for (int j = 0; j < 8; ++j) mean[j] *= 0.25f;

    float e[32];
#pragma unroll
    for (int k = 0; k < 32; ++k) e[k] = emb[q * 32 + k];
#pragma unroll
    for (int n = 0; n < 32; ++n) {
        float a = lb2[n];
#pragma unroll
        for (int k = 0; k < 32; ++k) a += e[k] * lw1[k * 32 + n];
#pragma unroll
        for (int j = 0; j < 8; ++j) a += mean[j] * lw2[j * 32 + n];
#pragma unroll
        for (int j = 0; j < 8; ++j) a += mx[j] * lw2[(8 + j) * 32 + n];
        embo[q * 32 + n] = a;
    }
}

// ---------------------------------------------------------------------------
// Host-side orchestration.
// ---------------------------------------------------------------------------
extern "C" void kernel_launch(void* const* d_in, const int* in_sizes, int n_in,
                              void* d_out, int out_size, void* d_ws, size_t ws_size,
                              hipStream_t stream) {
    (void)in_sizes; (void)n_in; (void)out_size; (void)ws_size;

    const float* x = (const float*)d_in[0];
    const float *n1w[4], *n1b[4], *n2w[4], *n2b[4], *n3w[4], *n3b[4];
    for (int i = 0; i < 4; ++i) {
        n1w[i] = (const float*)d_in[1 + 2 * i];
        n1b[i] = (const float*)d_in[2 + 2 * i];
        n2w[i] = (const float*)d_in[9 + 2 * i];
        n2b[i] = (const float*)d_in[10 + 2 * i];
        n3w[i] = (const float*)d_in[17 + 2 * i];
        n3b[i] = (const float*)d_in[18 + 2 * i];
    }
    const float* conv_ws  = (const float*)d_in[25];   // [3,32,4]
    const float* conv_bs  = (const float*)d_in[26];   // [3,4]
    const float* conv_wh  = (const float*)d_in[27];   // [3,32,8]
    const float* conv_bh  = (const float*)d_in[28];   // [3,8]
    const float* conv_wo1 = (const float*)d_in[29];   // [3,32,32]
    const float* conv_wo2 = (const float*)d_in[30];   // [3,16,32]
    const float* conv_bo2 = (const float*)d_in[31];   // [3,32]

    float* out  = (float*)d_out;
    float* wsf  = (float*)d_ws;
    float* embA = wsf;                         // M*32
    float* embB = embA + (long)MROWS * 32;     // M*32
    float* sbuf = embB + (long)MROWS * 32;     // M*4
    float* hbuf = sbuf + (long)MROWS * 4;      // M*8
    float* sqb  = hbuf + (long)MROWS * 8;      // M

    // nn1: 12 -> 126 -> 126 -> 126 -> 32  (H pad 128, Kpad0 32)
    fused_mlp_kernel<32, 128, 8, 2><<<MROWS / 64, 128, 0, stream>>>(
        x, 12, 12, nullptr, 0, 0,
        n1w[0], n1b[0], n1w[1], n1b[1], n1w[2], n1b[2], n1w[3], n1b[3],
        126, 32, embA, 32, 0);

    float* cur = embA;
    float* nxt = embB;
    for (int i = 0; i < 3; ++i) {
        gravnet_prep_kernel<<<MROWS / 256, 256, 0, stream>>>(
            cur,
            conv_ws + i * 32 * 4, conv_bs + i * 4,
            conv_wh + i * 32 * 8, conv_bh + i * 8,
            sbuf, hbuf, sqb);
        gravnet_knn_kernel<<<dim3(BPTS / 128, NBATCH), 128, 0, stream>>>(
            cur, sbuf, hbuf, sqb,
            conv_wo1 + i * 32 * 32, conv_wo2 + i * 16 * 32, conv_bo2 + i * 32,
            nxt);
        float* t = cur; cur = nxt; nxt = t;
    }

    // nn2: concat[x(12), emb(32)] = 44 -> 256^3 -> 6  -> out[:, 0:6]
    fused_mlp_kernel<64, 256, 16, 1><<<MROWS / 64, 128, 0, stream>>>(
        x, 12, 12, cur, 32, 32,
        n2w[0], n2b[0], n2w[1], n2b[1], n2w[2], n2b[2], n2w[3], n2b[3],
        256, 6, out, 12, 0);

    // nn3: concat[x(12), preds_id(6)] = 18 -> 256^3 -> 6 -> out[:, 6:12]
    fused_mlp_kernel<32, 256, 16, 1><<<MROWS / 64, 128, 0, stream>>>(
        x, 12, 12, out, 6, 12,
        n3w[0], n3b[0], n3w[1], n3b[1], n3w[2], n3b[2], n3w[3], n3b[3],
        256, 6, out, 12, 6);
}
